// RNN_12661563588811
// MI455X (gfx1250) — compile-verified, matrix operand order checked by reading, not run on hardware
//
#include <hip/hip_runtime.h>
#include <hip/hip_bf16.h>
#include <math.h>

// ---------------------------------------------------------------------------
// GRU recurrence on MI455X (gfx1250, wave32, WMMA).
//   states[B,T,H], h_n[B,H], mode_probs[B,T,M] = GRU(input[B,T,I], h_0, ...)
// Matrix path: v_wmma_f32_16x16x32_bf16 (f32 accumulate).
// All operands pre-converted to bf16 off the critical path; the 512-step
// dependent chain does only b128 loads -> WMMA -> LDS reduce -> gate VALU.
// ---------------------------------------------------------------------------

typedef __bf16 bf16_t;
typedef __attribute__((ext_vector_type(16))) __bf16 v16bf;
typedef __attribute__((ext_vector_type(8)))  __bf16 v8bf;
typedef __attribute__((ext_vector_type(8)))  float  v8f;

#define NB 64
#define NT 512
#define NI 512
#define NH 1024
#define NM 8
#define G3H (3 * NH)

// ---- WMMA fragment loader (wave32, 16x16x32 bf16) -------------------------
// Both A and B fragments: lane holds one row/col, 8 contiguous bf16 at kb
// and 8 more at kb+16 (kb = k0 + (lane>>4)*8). Two 16-byte loads.
__device__ __forceinline__ v16bf load_frag(const bf16_t* __restrict__ p) {
  const v8bf lo = *(const v8bf*)(p);        // K = kb .. kb+7
  const v8bf hi = *(const v8bf*)(p + 16);   // K = kb+16 .. kb+23
  v16bf f;
#pragma unroll
  for (int i = 0; i < 8; ++i) { f[i] = lo[i]; f[i + 8] = hi[i]; }
  return f;
}

__device__ __forceinline__ v8f wmma_bf16(v16bf a, v16bf b, v8f c) {
  return __builtin_amdgcn_wmma_f32_16x16x32_bf16(
      /*neg_a=*/false, a, /*neg_b=*/false, b,
      /*c_mod=*/(short)0, c, /*reuse_a=*/false, /*reuse_b=*/false);
}

// ---- one-time weight conversion: fp32 [K,3H] -> bf16 transposed [3H,K] ----
__global__ void convert_wT(const float* __restrict__ W, bf16_t* __restrict__ WT,
                           int K /*rows of W*/) {
  const long total = (long)K * G3H;
  for (long idx = (long)blockIdx.x * blockDim.x + threadIdx.x; idx < total;
       idx += (long)gridDim.x * blockDim.x) {
    const long n = idx / K;   // output channel (0..3H-1)
    const long k = idx % K;   // contraction index
    WT[idx] = (__bf16)W[k * G3H + n];
  }
}

// ---- one-time fp32 -> bf16 flat conversion (input tensor, h_0) ------------
__global__ void convert_bf16(const float* __restrict__ src,
                             bf16_t* __restrict__ dst, long n) {
  for (long idx = (long)blockIdx.x * blockDim.x + threadIdx.x; idx < n;
       idx += (long)gridDim.x * blockDim.x)
    dst[idx] = (__bf16)src[idx];
}

// ---- one GRU time step -----------------------------------------------------
// grid (4, 64): 16x16 tile of [B=64, H=1024]. block = 128 (4 waves).
// Waves split K=1536 (x:512 + h:1024); z/r accumulate x+h jointly, the n-gate
// keeps xn / hn separate (n = tanh(xn + r*hn)).
__global__ __launch_bounds__(128) void gru_step(
    const bf16_t* __restrict__ xb,    // [B,T,I]  bf16
    const bf16_t* __restrict__ hb_in, // [B,H]    bf16 (previous h)
    const float* __restrict__ hf_in,  // [B,H]    fp32 (previous h)
    const bf16_t* __restrict__ WxT,   // [3H,I] bf16 transposed
    const bf16_t* __restrict__ WhT,   // [3H,H] bf16 transposed
    const float* __restrict__ bias,   // [3H]  packed [z|r|n]
    bf16_t* __restrict__ hb_out,      // [B,H] bf16
    float* __restrict__ hf_out,       // [B,H] fp32
    float* __restrict__ states,       // [B,T,H]
    int t) {
  const int bx = blockIdx.x;          // b-tile (0..3)
  const int by = blockIdx.y;          // h-tile (0..63)
  const int tid = threadIdx.x;
  const int wave = tid >> 5;          // 0..3
  const int lane = tid & 31;
  const int mrow = lane & 15;
  const int khalf = (lane >> 4) * 8;  // 0 or 8

  const int brow = bx * 16 + mrow;                 // batch row for A frags
  const long jz = (long)0 * NH + by * 16 + mrow;   // B-frag col == lane&15
  const long jr = (long)1 * NH + by * 16 + mrow;
  const long jn = (long)2 * NH + by * 16 + mrow;

  v8f cz = {}, cr = {}, cxn = {}, chn = {};

  // -------- phase X: K over I=512, this wave owns a 128-wide K chunk -------
  {
    const bf16_t* xrow = xb + ((long)brow * NT + t) * NI;
    const bf16_t* wz = WxT + jz * NI;
    const bf16_t* wr = WxT + jr * NI;
    const bf16_t* wn = WxT + jn * NI;
#pragma unroll
    for (int kk = 0; kk < 4; ++kk) {
      const int kb = wave * 128 + kk * 32 + khalf;
      const v16bf a  = load_frag(xrow + kb);
      const v16bf bz = load_frag(wz + kb);
      const v16bf br = load_frag(wr + kb);
      const v16bf bn = load_frag(wn + kb);
      cz  = wmma_bf16(a, bz, cz);
      cr  = wmma_bf16(a, br, cr);
      cxn = wmma_bf16(a, bn, cxn);
    }
  }
  // -------- phase H: K over H=1024, this wave owns a 256-wide K chunk ------
  {
    const bf16_t* hrow = hb_in + (long)brow * NH;
    const bf16_t* wz = WhT + jz * NH;
    const bf16_t* wr = WhT + jr * NH;
    const bf16_t* wn = WhT + jn * NH;
#pragma unroll
    for (int kk = 0; kk < 8; ++kk) {
      const int kb = wave * 256 + kk * 32 + khalf;
      const v16bf a  = load_frag(hrow + kb);
      const v16bf bz = load_frag(wz + kb);
      const v16bf br = load_frag(wr + kb);
      const v16bf bn = load_frag(wn + kb);
      cz  = wmma_bf16(a, bz, cz);
      cr  = wmma_bf16(a, br, cr);
      chn = wmma_bf16(a, bn, chn);
    }
  }

  // -------- cross-wave K reduction via LDS (v8f -> paired ds_*_b128) -------
  __shared__ v8f red[3][4][32];  // [wave-1][acc][lane] : 12 KB
  if (wave > 0) {
    red[wave - 1][0][lane] = cz;
    red[wave - 1][1][lane] = cr;
    red[wave - 1][2][lane] = cxn;
    red[wave - 1][3][lane] = chn;
  }
  __syncthreads();
  if (wave == 0) {
#pragma unroll
    for (int w = 0; w < 3; ++w) {
      cz  += red[w][0][lane];
      cr  += red[w][1][lane];
      cxn += red[w][2][lane];
      chn += red[w][3][lane];
    }
    // -------- fused gate math + state update (C row = v + 8*(lane>>4)) -----
    const int j = by * 16 + (lane & 15);
#pragma unroll
    for (int v = 0; v < 8; ++v) {
      const int m  = v + 8 * (lane >> 4);
      const int bg = bx * 16 + m;
      const float z = 1.0f / (1.0f + __expf(-(cz[v] + bias[j])));
      const float r = 1.0f / (1.0f + __expf(-(cr[v] + bias[NH + j])));
      const float n = tanhf(cxn[v] + bias[2 * NH + j] + r * chn[v]);
      const float hold = hf_in[(long)bg * NH + j];
      const float hnew = (1.0f - z) * hold + z * n;
      hf_out[(long)bg * NH + j] = hnew;
      hb_out[(long)bg * NH + j] = (__bf16)hnew;
      states[((long)bg * NT + t) * NH + j] = hnew;
    }
  }
}

// ---- h_n = final hidden state ---------------------------------------------
__global__ void copy_hn(const float* __restrict__ hfin, float* __restrict__ hn) {
  const int idx = blockIdx.x * blockDim.x + threadIdx.x;
  if (idx < NB * NH) hn[idx] = hfin[idx];
}

// ---- mode head: softmax(states @ Wm + bm), one wave per (b,t) row ----------
__global__ __launch_bounds__(256) void mode_head(
    const float* __restrict__ states,  // [B*T, H]
    const float* __restrict__ Wm,      // [H, M]
    const float* __restrict__ bm,      // [M]
    float* __restrict__ mp) {          // [B*T, M]
  const int row = blockIdx.x * 8 + (threadIdx.x >> 5);
  const int lane = threadIdx.x & 31;
  if (row >= NB * NT) return;
  const float* s = states + (long)row * NH;

  float acc[NM];
#pragma unroll
  for (int m = 0; m < NM; ++m) acc[m] = 0.0f;
  for (int k = lane; k < NH; k += 32) {
    const float sv = s[k];
    const float* w = Wm + (long)k * NM;
#pragma unroll
    for (int m = 0; m < NM; ++m) acc[m] += sv * w[m];
  }
#pragma unroll
  for (int off = 16; off > 0; off >>= 1) {
#pragma unroll
    for (int m = 0; m < NM; ++m) acc[m] += __shfl_xor(acc[m], off, 32);
  }
  // every lane now holds all 8 logits
  float logit[NM], mx = -3.4e38f;
#pragma unroll
  for (int m = 0; m < NM; ++m) {
    logit[m] = acc[m] + bm[m];
    mx = fmaxf(mx, logit[m]);
  }
  float denom = 0.0f;
#pragma unroll
  for (int m = 0; m < NM; ++m) {
    logit[m] = __expf(logit[m] - mx);
    denom += logit[m];
  }
  const float inv = 1.0f / denom;
  if (lane < NM) mp[(long)row * NM + lane] = logit[lane] * inv;
}

// ---------------------------------------------------------------------------
extern "C" void kernel_launch(void* const* d_in, const int* in_sizes, int n_in,
                              void* d_out, int out_size, void* d_ws,
                              size_t ws_size, hipStream_t stream) {
  const float* input = (const float*)d_in[0];  // [B,T,I]
  const float* h0    = (const float*)d_in[1];  // [B,H]
  const float* Wx    = (const float*)d_in[2];  // [I,3H]
  const float* Wh    = (const float*)d_in[3];  // [H,3H]
  const float* bias  = (const float*)d_in[4];  // [3H]
  const float* Wm    = (const float*)d_in[5];  // [H,M]
  const float* bm    = (const float*)d_in[6];  // [M]
  (void)in_sizes; (void)n_in; (void)out_size; (void)ws_size;

  // d_out: states [B,T,H] | h_n [B,H] | mode_probs [B,T,M]
  float* states = (float*)d_out;
  float* hn_out = states + (long)NB * NT * NH;
  float* mp_out = hn_out + (long)NB * NH;

  // workspace carve (~44 MB):
  //   bf16 transposed weights | bf16 input | bf16 h0 | bf16 h ping-pong |
  //   fp32 h ping-pong
  char* ws = (char*)d_ws;
  size_t off = 0;
  bf16_t* WxT = (bf16_t*)(ws + off); off += (size_t)G3H * NI * 2;
  bf16_t* WhT = (bf16_t*)(ws + off); off += (size_t)G3H * NH * 2;
  bf16_t* Xb  = (bf16_t*)(ws + off); off += (size_t)NB * NT * NI * 2;
  bf16_t* h0b = (bf16_t*)(ws + off); off += (size_t)NB * NH * 2;
  bf16_t* hb0 = (bf16_t*)(ws + off); off += (size_t)NB * NH * 2;
  bf16_t* hb1 = (bf16_t*)(ws + off); off += (size_t)NB * NH * 2;
  float*  hf0 = (float*)(ws + off);  off += (size_t)NB * NH * 4;
  float*  hf1 = (float*)(ws + off);  off += (size_t)NB * NH * 4;

  // one-time (per call) conversions -- fully parallel, off the critical path
  convert_wT<<<4096, 256, 0, stream>>>(Wx, WxT, NI);
  convert_wT<<<4096, 256, 0, stream>>>(Wh, WhT, NH);
  convert_bf16<<<8192, 256, 0, stream>>>(input, Xb, (long)NB * NT * NI);
  convert_bf16<<<256, 256, 0, stream>>>(h0, h0b, (long)NB * NH);

  // sequential recurrence: 512 dependent step kernels, h ping-pong
  const dim3 grid(NB / 16, NH / 16);
  const bf16_t* hb_read = h0b;
  const float*  hf_read = h0;
  bf16_t* hbufs_b[2] = {hb0, hb1};
  float*  hbufs_f[2] = {hf0, hf1};
  for (int t = 0; t < NT; ++t) {
    bf16_t* hb_write = hbufs_b[t & 1];
    float*  hf_write = hbufs_f[t & 1];
    gru_step<<<grid, 128, 0, stream>>>(Xb, hb_read, hf_read, WxT, WhT, bias,
                                       hb_write, hf_write, states, t);
    hb_read = hb_write;
    hf_read = hf_write;
  }

  copy_hn<<<(NB * NH + 255) / 256, 256, 0, stream>>>(hf_read, hn_out);
  mode_head<<<(NB * NT) / 8, 256, 0, stream>>>(states, Wm, bm, mp_out);
}
